// EncoderSaliencySelection_7705171329184
// MI455X (gfx1250) — matrix operand order repaired and checked by reading, not compiled, not run on hardware
//
#include <hip/hip_runtime.h>
#include <hip/hip_bf16.h>
#include <math.h>

// ---------------------------------------------------------------------------
// Problem constants (from reference)
// ---------------------------------------------------------------------------
#define B_    32
#define N_    4096
#define IN_   125
#define AD_   128      // anchor dim (padded input 125 + 3 features)
#define HID_  64
#define KD_   64       // lift output dim
#define DM_   1024     // d_model
#define KSEL_ 256      // top-k
// LAM = 0.5 -> softmax(sal * 2)

typedef __attribute__((ext_vector_type(16))) __bf16 v16bf;
typedef __attribute__((ext_vector_type(8)))  __bf16 v8bf;
typedef __attribute__((ext_vector_type(8)))  float  v8f;

__device__ inline __bf16 f2bf(float f) {
    unsigned int u = __float_as_uint(f);
    unsigned int r = (u + 0x7FFFu + ((u >> 16) & 1u)) >> 16;
    unsigned short s = (unsigned short)r;
    __bf16 out;
    __builtin_memcpy(&out, &s, 2);
    return out;
}

// ---------------------------------------------------------------------------
// WMMA bf16 16x16x32 fragments (layouts per CDNA5 ISA 7.12.2)
//
// A (16xK, row-major bf16 in LDS, 16B-aligned rows): lane = hh*16+lr owns
// row M=lr; its 16 elements are two contiguous 8-elem runs:
//   K = kc*32 + hh*8 + {0..7}   and   K = kc*32 + 16 + hh*8 + {0..7}
// -> two ds_load_b128.
//
// B (KxN): pre-packed in LDS at staging time. Fragment f = kc*NT + nt is a
// [32 lanes][16 bf16] contiguous block; lane data = column (nt*16+lr),
// K = kc*32 + hh*8 + {0..7}, kc*32 + 16 + hh*8 + {0..7}  -> two ds_load_b128.
//
// C/D f32: VGPR r -> (M = r + 8*hh, N = lr)
// ---------------------------------------------------------------------------
__device__ inline v16bf load_a_frag(const __bf16* base, int stride,
                                    int rowbase, int kc, int lane) {
    int hh = lane >> 4, lr = lane & 15;
    const __bf16* rp = base + (size_t)(rowbase + lr) * stride + kc * 32 + hh * 8;
    v8bf lo = *(const v8bf*)(rp);
    v8bf hi = *(const v8bf*)(rp + 16);
    return __builtin_shufflevector(lo, hi, 0, 1, 2, 3, 4, 5, 6, 7,
                                           8, 9, 10, 11, 12, 13, 14, 15);
}

__device__ inline v16bf load_b_packed(const __bf16* base, int NT,
                                      int kc, int nt, int lane) {
    const v8bf* p =
        (const v8bf*)(base + (size_t)(((kc * NT + nt) * 32) + lane) * 16);
    v8bf lo = p[0], hi = p[1];
    return __builtin_shufflevector(lo, hi, 0, 1, 2, 3, 4, 5, 6, 7,
                                           8, 9, 10, 11, 12, 13, 14, 15);
}

// Stage global f32 weight W[k, c] (row stride gstride, starting col c0) into
// packed-fragment LDS buffer: KC k-chunks x NT 16-col tiles. Rows >= valid_k
// are zero-padded. Coalesced: adjacent threads take adjacent lanes -> cols.
__device__ inline void stage_b_packed(__bf16* dst, const float* __restrict__ W,
                                      int gstride, int c0, int KC, int NT,
                                      int valid_k, int tid, int nthreads) {
    int total = KC * NT * 32;                 // fragment-lanes
    for (int fl = tid; fl < total; fl += nthreads) {
        int lane = fl & 31, f = fl >> 5;
        int kc = f / NT, nt = f - kc * NT;
        int hh = lane >> 4, lr = lane & 15;
        int col = nt * 16 + lr;
        int kbase = kc * 32 + hh * 8;
        v8bf lo, hi;
#pragma unroll
        for (int kk = 0; kk < 8; ++kk) {
            int k0 = kbase + kk, k1 = kbase + 16 + kk;
            float v0 = (k0 < valid_k) ? W[(size_t)k0 * gstride + c0 + col] : 0.f;
            float v1 = (k1 < valid_k) ? W[(size_t)k1 * gstride + c0 + col] : 0.f;
            lo[kk] = f2bf(v0);
            hi[kk] = f2bf(v1);
        }
        v8bf* out = (v8bf*)(dst + (size_t)fl * 16);
        out[0] = lo;
        out[1] = hi;
    }
}

#define WMMA_BF16(A, Bf, C)                                                   \
    __builtin_amdgcn_wmma_f32_16x16x32_bf16(false, (A), false, (Bf),          \
                                            (short)0, (C), false, false)

// ---------------------------------------------------------------------------
// Kernel 1: saliency = sigmoid( relu(x@W1+b1) @ W2 + b2 )
// 256 threads (8 waves), 128 points per block. Each wave owns a 16-point tile.
// ---------------------------------------------------------------------------
__global__ __launch_bounds__(256)
void k_saliency(const float* __restrict__ x,  const float* __restrict__ W1,
                const float* __restrict__ b1, const float* __restrict__ W2,
                const float* __restrict__ b2, float* __restrict__ sal) {
    __shared__ alignas(16) __bf16 sA[128 * 128];      // 32 KB, zero-padded cols
    __shared__ alignas(16) __bf16 sW1p[4 * 4 * 32 * 16]; // packed frags, 16 KB
    __shared__ float sW2[64];
    __shared__ float sB1[64];

    int tid = threadIdx.x;
    int gp0 = blockIdx.x * 128;        // flat (b*N+n) start; block stays in one b

    for (int i = tid; i < 128 * 128; i += 256) {
        int r = i >> 7, c = i & 127;
        float v = (c < IN_) ? x[(size_t)(gp0 + r) * IN_ + c] : 0.f;
        sA[i] = f2bf(v);
    }
    stage_b_packed(sW1p, W1, HID_, 0, /*KC=*/4, /*NT=*/4, IN_, tid, 256);
    if (tid < 64) { sW2[tid] = W2[tid]; sB1[tid] = b1[tid]; }
    __syncthreads();

    int wave = tid >> 5, lane = tid & 31, hh = lane >> 4, lr = lane & 15;
    int rowbase = wave * 16;

    v8f zero = {};
    v8f acc[4];
#pragma unroll
    for (int i = 0; i < 4; ++i) acc[i] = zero;

#pragma unroll
    for (int kc = 0; kc < 4; ++kc) {
        v16bf a = load_a_frag(sA, 128, rowbase, kc, lane);
#pragma unroll
        for (int nt = 0; nt < 4; ++nt) {
            v16bf bf = load_b_packed(sW1p, 4, kc, nt, lane);
            acc[nt] = WMMA_BF16(a, bf, acc[nt]);
        }
    }

    float b2v = b2[0];
#pragma unroll
    for (int r = 0; r < 8; ++r) {
        float part = 0.f;
#pragma unroll
        for (int nt = 0; nt < 4; ++nt) {
            int j = nt * 16 + lr;
            float h = acc[nt][r] + sB1[j];
            h = fmaxf(h, 0.f);
            part += h * sW2[j];
        }
        // reduce across the 16-lane half-group (masks < 16 stay in group)
        part += __shfl_xor(part, 1);
        part += __shfl_xor(part, 2);
        part += __shfl_xor(part, 4);
        part += __shfl_xor(part, 8);
        if (lr == r) {
            float s = part + b2v;
            int m = rowbase + hh * 8 + r;
            sal[(size_t)gp0 + m] = 1.f / (1.f + expf(-s));
        }
    }
}

// ---------------------------------------------------------------------------
// Kernel 2: per-batch softmax (y_star), cumsum/N (csal), top-256 indices via
// in-LDS bitonic sort of packed keys (sal bits desc, index asc).
// One 1024-thread block per batch.
// ---------------------------------------------------------------------------
__global__ __launch_bounds__(1024)
void k_batchstats(const float* __restrict__ sal, float* __restrict__ csal,
                  int* __restrict__ topidx, float* __restrict__ ystar) {
    __shared__ float              sS[N_];     // 16 KB
    __shared__ unsigned long long sK[N_];     // 32 KB
    __shared__ float              sR[1024];   // 4 KB

    int b = blockIdx.x, tid = threadIdx.x;
    const float* sb = sal + (size_t)b * N_;

    for (int i = tid; i < N_; i += 1024) sS[i] = sb[i];
    __syncthreads();

    // --- max reduce ---
    float m = -1e30f;
    for (int i = tid; i < N_; i += 1024) m = fmaxf(m, sS[i]);
    sR[tid] = m; __syncthreads();
    for (int off = 512; off > 0; off >>= 1) {
        if (tid < off) sR[tid] = fmaxf(sR[tid], sR[tid + off]);
        __syncthreads();
    }
    float mx2 = sR[0] * 2.f;                  // softmax over sal*2 (LAM=0.5)
    __syncthreads();

    // --- sum exp ---
    float s = 0.f;
    for (int i = tid; i < N_; i += 1024) s += expf(sS[i] * 2.f - mx2);
    sR[tid] = s; __syncthreads();
    for (int off = 512; off > 0; off >>= 1) {
        if (tid < off) sR[tid] += sR[tid + off];
        __syncthreads();
    }
    float inv = 1.f / sR[0];
    __syncthreads();

    for (int i = tid; i < N_; i += 1024)
        ystar[(size_t)b * N_ + i] = expf(sS[i] * 2.f - mx2) * inv;

    // --- cumsum (blocked 4/thread + Hillis-Steele scan of thread totals) ---
    float e0 = sS[tid * 4], e1 = sS[tid * 4 + 1];
    float e2 = sS[tid * 4 + 2], e3 = sS[tid * 4 + 3];
    float p1 = e0 + e1, p2 = p1 + e2, p3 = p2 + e3;
    sR[tid] = p3; __syncthreads();
    for (int off = 1; off < 1024; off <<= 1) {
        float v = (tid >= off) ? sR[tid - off] : 0.f;
        __syncthreads();
        sR[tid] += v;
        __syncthreads();
    }
    float excl = sR[tid] - p3;
    const float invN = 1.f / (float)N_;
    size_t cbase = (size_t)b * N_ + tid * 4;
    csal[cbase + 0] = (excl + e0) * invN;
    csal[cbase + 1] = (excl + p1) * invN;
    csal[cbase + 2] = (excl + p2) * invN;
    csal[cbase + 3] = (excl + p3) * invN;
    __syncthreads();

    // --- bitonic sort of (sal, index): ascending; read top-k from the tail.
    // key = bits(sal)<<32 | (4095-n): equal sal -> larger (4095-n) later in
    // ascending order -> smaller n first when reading from the end (JAX tie rule)
    for (int i = tid; i < N_; i += 1024) {
        unsigned int bits = __float_as_uint(sS[i]);   // sigmoid > 0 -> monotone
        sK[i] = ((unsigned long long)bits << 32) |
                (unsigned long long)(unsigned int)(N_ - 1 - i);
    }
    __syncthreads();
    for (int k = 2; k <= N_; k <<= 1) {
        for (int j = k >> 1; j > 0; j >>= 1) {
            for (int i = tid; i < N_; i += 1024) {
                int ixj = i ^ j;
                if (ixj > i) {
                    unsigned long long a = sK[i], c = sK[ixj];
                    bool up = ((i & k) == 0);
                    if (up ? (a > c) : (a < c)) { sK[i] = c; sK[ixj] = a; }
                }
            }
            __syncthreads();
        }
    }
    if (tid < KSEL_) {
        unsigned long long key = sK[N_ - 1 - tid];
        int n = (N_ - 1) - (int)(unsigned int)(key & 0xFFFFFFFFull);
        topidx[b * KSEL_ + tid] = n;
    }
}

// ---------------------------------------------------------------------------
// Kernel 3: for the 256 selected points only: build normalized anchors,
// lift 128->64 with WMMA. 128 points per block (grid = B*2).
// ---------------------------------------------------------------------------
__global__ __launch_bounds__(256)
void k_cloud(const float* __restrict__ x,     const float* __restrict__ liftW,
             const float* __restrict__ liftb, const float* __restrict__ mu,
             const float* __restrict__ sigma, const float* __restrict__ sal,
             const float* __restrict__ csal,  const int* __restrict__ topidx,
             float* __restrict__ cloud) {
    __shared__ alignas(16) __bf16 sD[128 * AD_];         // 32 KB anchors
    __shared__ alignas(16) __bf16 sLWp[4 * 4 * 32 * 16]; // packed frags, 16 KB
    __shared__ float sLB[KD_];

    int b = blockIdx.x >> 1;
    int prange = (blockIdx.x & 1) * 128;
    int tid = threadIdx.x;

    stage_b_packed(sLWp, liftW, KD_, 0, /*KC=*/4, /*NT=*/4, AD_, tid, 256);
    if (tid < KD_) sLB[tid] = liftb[tid];

    if (tid < 128) {
        int n = topidx[b * KSEL_ + prange + tid];
        const float* xr = x + ((size_t)b * N_ + n) * IN_;
        float sv = sal[(size_t)b * N_ + n];
        float cv = csal[(size_t)b * N_ + n];
        float pv = (float)n / (float)(N_ - 1);
        float ss = sv * sv + pv * pv + cv * cv;
        for (int d = 0; d < IN_; ++d) { float v = xr[d]; ss += v * v; }
        float invn = 1.f / (sqrtf(ss) + 1e-6f);
        __bf16* dr = sD + (size_t)tid * AD_;
        for (int d = 0; d < IN_; ++d)
            dr[d] = f2bf((xr[d] * invn - mu[d]) / sigma[d]);
        dr[125] = f2bf((sv * invn - mu[125]) / sigma[125]);
        dr[126] = f2bf((pv * invn - mu[126]) / sigma[126]);
        dr[127] = f2bf((cv * invn - mu[127]) / sigma[127]);
    }
    __syncthreads();

    int wave = tid >> 5, lane = tid & 31, hh = lane >> 4, lr = lane & 15;
    int rowbase = wave * 16;               // 8 waves x 16 rows = 128 points

    v8f zero = {};
    v8f acc[4];
#pragma unroll
    for (int i = 0; i < 4; ++i) acc[i] = zero;

#pragma unroll
    for (int kc = 0; kc < 4; ++kc) {
        v16bf a = load_a_frag(sD, AD_, rowbase, kc, lane);
#pragma unroll
        for (int nt = 0; nt < 4; ++nt) {
            v16bf bf = load_b_packed(sLWp, 4, kc, nt, lane);
            acc[nt] = WMMA_BF16(a, bf, acc[nt]);
        }
    }
#pragma unroll
    for (int nt = 0; nt < 4; ++nt) {
        int col = nt * 16 + lr;
#pragma unroll
        for (int r = 0; r < 8; ++r) {
            int row = prange + rowbase + hh * 8 + r;
            cloud[((size_t)b * KSEL_ + row) * KD_ + col] = acc[nt][r] + sLB[col];
        }
    }
}

// ---------------------------------------------------------------------------
// Kernel 4: tokens = cloud[256,64] @ proj_W[64,1024] + proj_b
// grid = B*8 blocks; each block does all 256 rows x 128 columns.
// ---------------------------------------------------------------------------
__global__ __launch_bounds__(256)
void k_proj(const float* __restrict__ cloud, const float* __restrict__ projW,
            const float* __restrict__ projb, float* __restrict__ tokens) {
    __shared__ alignas(16) __bf16 sC[KSEL_ * KD_];       // 32 KB (A, row-major)
    __shared__ alignas(16) __bf16 sPWp[2 * 8 * 32 * 16]; // packed frags, 16 KB
    __shared__ float sPB[128];

    int b = blockIdx.x >> 3, c0 = (blockIdx.x & 7) * 128;
    int tid = threadIdx.x;

    for (int i = tid; i < KSEL_ * KD_; i += 256)
        sC[i] = f2bf(cloud[(size_t)b * KSEL_ * KD_ + i]);
    stage_b_packed(sPWp, projW, DM_, c0, /*KC=*/2, /*NT=*/8, KD_, tid, 256);
    if (tid < 128) sPB[tid] = projb[c0 + tid];
    __syncthreads();

    int wave = tid >> 5, lane = tid & 31, hh = lane >> 4, lr = lane & 15;
    v8f zero = {};

#pragma unroll
    for (int mi = 0; mi < 2; ++mi) {
        int rowbase = (wave * 2 + mi) * 16;
#pragma unroll
        for (int nt = 0; nt < 8; ++nt) {
            v8f acc = zero;
#pragma unroll
            for (int kc = 0; kc < 2; ++kc) {
                v16bf a  = load_a_frag(sC, KD_, rowbase, kc, lane);
                v16bf bf = load_b_packed(sPWp, 8, kc, nt, lane);
                acc = WMMA_BF16(a, bf, acc);
            }
            int col = nt * 16 + lr;
#pragma unroll
            for (int r = 0; r < 8; ++r) {
                int row = rowbase + hh * 8 + r;
                tokens[((size_t)b * KSEL_ + row) * DM_ + c0 + col] =
                    acc[r] + sPB[col];
            }
        }
    }
}

// ---------------------------------------------------------------------------
// Host launcher
// ---------------------------------------------------------------------------
extern "C" void kernel_launch(void* const* d_in, const int* in_sizes, int n_in,
                              void* d_out, int out_size, void* d_ws,
                              size_t ws_size, hipStream_t stream) {
    const float* x     = (const float*)d_in[0];
    const float* W1    = (const float*)d_in[1];
    const float* b1    = (const float*)d_in[2];
    const float* W2    = (const float*)d_in[3];
    const float* b2    = (const float*)d_in[4];
    const float* liftW = (const float*)d_in[5];
    const float* liftb = (const float*)d_in[6];
    const float* mu    = (const float*)d_in[7];
    const float* sigma = (const float*)d_in[8];
    const float* projW = (const float*)d_in[9];
    const float* projb = (const float*)d_in[10];

    char* ws = (char*)d_ws;
    float* sal    = (float*)(ws);                                  // B*N f32
    float* csal   = (float*)(ws + (size_t)B_ * N_ * 4);            // B*N f32
    int*   topidx = (int*)  (ws + (size_t)2 * B_ * N_ * 4);        // B*256 i32
    float* cloud  = (float*)(ws + (size_t)2 * B_ * N_ * 4 +
                                  (size_t)B_ * KSEL_ * 4);         // B*256*64

    float* tokens = (float*)d_out;
    float* ystar  = (float*)d_out + (size_t)B_ * KSEL_ * DM_;

    k_saliency  <<<(B_ * N_) / 128, 256, 0, stream>>>(x, W1, b1, W2, b2, sal);
    k_batchstats<<<B_, 1024, 0, stream>>>(sal, csal, topidx, ystar);
    k_cloud     <<<B_ * 2, 256, 0, stream>>>(x, liftW, liftb, mu, sigma,
                                             sal, csal, topidx, cloud);
    k_proj      <<<B_ * 8, 256, 0, stream>>>(cloud, projW, projb, tokens);
}
